// CausalSelfAttention_89197880803915
// MI455X (gfx1250) — matrix operand order repaired and checked by reading, not compiled, not run on hardware
//
#include <hip/hip_runtime.h>
#include <hip/hip_bf16.h>

// Problem constants (match reference)
constexpr int Bc = 4;
constexpr int Tc = 2048;
constexpr int Cc = 1024;
constexpr int Hc = 16;
constexpr int Dc = 64;
constexpr int Mc = Bc * Tc; // 8192 tokens

typedef __bf16 bf16_t;
typedef __attribute__((ext_vector_type(16))) __bf16 v16bf;
typedef __attribute__((ext_vector_type(8)))  float  v8f;

// fp32 -> bf16 round-to-nearest-even
__device__ __forceinline__ bf16_t f2bf(float f) {
  unsigned u = __builtin_bit_cast(unsigned, f);
  u = (u + 0x7FFFu + ((u >> 16) & 1u)) >> 16;
  return __builtin_bit_cast(bf16_t, (unsigned short)u);
}

union frag_u { v16bf v; uint4 q[2]; };

// A-operand (16x32 bf16) per ISA layout: lane row = lane&15,
// K chunks [khalf*8, +8) and [16+khalf*8, +8)  -> two 16B loads from a row.
__device__ __forceinline__ v16bf load_a_frag(const bf16_t* row, int khalf) {
  frag_u f;
  f.q[0] = *(const uint4*)(row + khalf * 8);
  f.q[1] = *(const uint4*)(row + 16 + khalf * 8);
  return f.v;
}

// B-operand (32x16 bf16) per ISA layout: lane col = lane&15,
// K chunks [khalf*8, +8) and [16+khalf*8, +8) -> two 16B loads (base = column data).
__device__ __forceinline__ v16bf load_b_frag(const bf16_t* base, int khalf) {
  frag_u f;
  f.q[0] = *(const uint4*)(base + khalf * 8);
  f.q[1] = *(const uint4*)(base + 16 + khalf * 8);
  return f.v;
}

// ---------------- conversion kernels ----------------

__global__ __launch_bounds__(256) void cvt_f32_to_bf16(
    const float* __restrict__ src, bf16_t* __restrict__ dst, int n) {
  int i = (blockIdx.x * 256 + threadIdx.x) * 4;
  if (i + 3 < n) {
    float4 f = *(const float4*)(src + i);
    dst[i + 0] = f2bf(f.x);
    dst[i + 1] = f2bf(f.y);
    dst[i + 2] = f2bf(f.z);
    dst[i + 3] = f2bf(f.w);
  }
}

// W [C,C] (k,n) fp32 -> WT [C,C] (n,k) bf16
__global__ __launch_bounds__(256) void cvt_transpose_bf16(
    const float* __restrict__ w, bf16_t* __restrict__ wT) {
  int idx = blockIdx.x * 256 + threadIdx.x; // over C*C
  int k = idx >> 10;
  int n = idx & (Cc - 1);
  wT[n * Cc + k] = f2bf(w[idx]);
}

// ---------------- shared GEMM helpers (32x64 wave tile) ----------------

// load one K-step's fragments: 2 A frags + 4 B frags
__device__ __forceinline__ void gemm_load_step(
    const bf16_t* arow0, const bf16_t* arow1, const bf16_t* wrow,
    int kk, int khalf, v16bf& a0, v16bf& a1, v16bf (&b)[4]) {
  a0 = load_a_frag(arow0 + kk, khalf);
  a1 = load_a_frag(arow1 + kk, khalf);
#pragma unroll
  for (int nt = 0; nt < 4; ++nt)
    b[nt] = load_b_frag(wrow + (size_t)nt * 16 * Cc + kk, khalf);
}

// 8 WMMAs: 2 M-subtiles x 4 N-subtiles, B fragments reused across M
__device__ __forceinline__ void gemm_wmma8(
    v8f (&acc)[2][4], const v16bf& a0, const v16bf& a1, const v16bf (&b)[4]) {
#pragma unroll
  for (int nt = 0; nt < 4; ++nt)
    acc[0][nt] = __builtin_amdgcn_wmma_f32_16x16x32_bf16(
        false, a0, false, b[nt], (short)0, acc[0][nt], false, false);
#pragma unroll
  for (int nt = 0; nt < 4; ++nt)
    acc[1][nt] = __builtin_amdgcn_wmma_f32_16x16x32_bf16(
        false, a1, false, b[nt], (short)0, acc[1][nt], false, false);
}

// ping-pong mainloop over K (Cc/32 steps, unrolled by 2: no buffer copies)
__device__ __forceinline__ void gemm_mainloop(
    const bf16_t* arow0, const bf16_t* arow1, const bf16_t* wrow,
    int khalf, v8f (&acc)[2][4]) {
  v16bf a0A, a1A, bA[4];
  v16bf a0B, a1B, bB[4];
  gemm_load_step(arow0, arow1, wrow, 0, khalf, a0A, a1A, bA);
  for (int kk = 0; kk < Cc - 64; kk += 64) {
    gemm_load_step(arow0, arow1, wrow, kk + 32, khalf, a0B, a1B, bB);
    gemm_wmma8(acc, a0A, a1A, bA);
    gemm_load_step(arow0, arow1, wrow, kk + 64, khalf, a0A, a1A, bA);
    gemm_wmma8(acc, a0B, a1B, bB);
  }
  gemm_load_step(arow0, arow1, wrow, Cc - 32, khalf, a0B, a1B, bB);
  gemm_wmma8(acc, a0A, a1A, bA);
  gemm_wmma8(acc, a0B, a1B, bB);
}

// ---------------- fused QKV projection ----------------
// grid: (C/64, M/32/8, 3), block 256 (8 waves); wave computes a 32x64 tile.
__global__ __launch_bounds__(256) void qkv_gemm_kernel(
    const bf16_t* __restrict__ xb,
    const bf16_t* __restrict__ wqT, const bf16_t* __restrict__ wkT,
    const bf16_t* __restrict__ wvT,
    bf16_t* __restrict__ qo, bf16_t* __restrict__ ko, bf16_t* __restrict__ vo) {
  const int lane  = threadIdx.x & 31;
  const int wave  = threadIdx.x >> 5;
  const int khalf = lane >> 4;
  const int l16   = lane & 15;
  const int proj  = blockIdx.z;
  const bf16_t* wT = (proj == 0) ? wqT : (proj == 1) ? wkT : wvT;

  const int mbase = (blockIdx.y * 8 + wave) * 32;
  const int nbase = blockIdx.x * 64;

  const bf16_t* arow0 = xb + (size_t)(mbase + l16) * Cc;
  const bf16_t* arow1 = xb + (size_t)(mbase + 16 + l16) * Cc;
  const bf16_t* wrow  = wT + (size_t)(nbase + l16) * Cc;

  v8f acc[2][4] = {{{}, {}, {}, {}}, {{}, {}, {}, {}}};
  gemm_mainloop(arow0, arow1, wrow, khalf, acc);

  // scatter-store q/k as [B,H,T,D], v transposed as [B,H,D,T]
  const int bb = mbase >> 11;          // token block never straddles T
  const int tb = mbase & (Tc - 1);
#pragma unroll
  for (int sub = 0; sub < 2; ++sub) {
    const int t0 = tb + sub * 16 + khalf * 8;
#pragma unroll
    for (int nt = 0; nt < 4; ++nt) {
      const int n = nbase + nt * 16 + l16;
      const int h = n >> 6;
      const int d = n & (Dc - 1);
      if (proj == 0) {
        bf16_t* dst = qo + ((size_t)(bb * Hc + h) * Tc + t0) * Dc + d;
#pragma unroll
        for (int r = 0; r < 8; ++r) dst[r * Dc] = f2bf(acc[sub][nt][r]);
      } else if (proj == 1) {
        bf16_t* dst = ko + ((size_t)(bb * Hc + h) * Tc + t0) * Dc + d;
#pragma unroll
        for (int r = 0; r < 8; ++r) dst[r * Dc] = f2bf(acc[sub][nt][r]);
      } else {
        bf16_t* dst = vo + ((size_t)(bb * Hc + h) * Dc + d) * Tc + t0;
#pragma unroll
        for (int r = 0; r < 8; ++r) dst[r] = f2bf(acc[sub][nt][r]);
      }
    }
  }
}

// helper: B-operand fragments of K^T for one 32-key tile
__device__ __forceinline__ void load_k_frags(const bf16_t* kh, int keybase,
                                             int l16, int khalf,
                                             v16bf (&kf)[2][2]) {
#pragma unroll
  for (int sub = 0; sub < 2; ++sub) {
    const bf16_t* krow = kh + (size_t)(keybase + sub * 16 + l16) * Dc;
    kf[sub][0] = load_b_frag(krow, khalf);
    kf[sub][1] = load_b_frag(krow + 32, khalf);
  }
}

// one 32-key flash step: uses kf (already loaded), loads kfn for nextbase.
__device__ __forceinline__ void flash_step(
    const int keybase, const int nextbase,
    const bf16_t* kh, const bf16_t* vh, bf16_t* pw,
    const int qbase, const int l16, const int khalf,
    const v16bf& qa0, const v16bf& qa1,
    v16bf (&kf)[2][2], v16bf (&kfn)[2][2],
    v8f (&o)[4], float (&mrow)[8], float (&lrow)[8]) {
  // issue current tile's V fragment loads (consumed after softmax)
  v16bf vf[4];
#pragma unroll
  for (int dt = 0; dt < 4; ++dt)
    vf[dt] = load_b_frag(vh + (size_t)(dt * 16 + l16) * Tc + keybase, khalf);

  // issue next tile's K fragment loads (consumed next step)
  load_k_frags(kh, nextbase, l16, khalf, kfn);

  // S = Q @ K^T for 32 keys: two 16x16 subtiles, each K=64 -> 2 WMMAs
  v8f s[2];
#pragma unroll
  for (int sub = 0; sub < 2; ++sub) {
    v8f z = {};
    z = __builtin_amdgcn_wmma_f32_16x16x32_bf16(false, qa0, false, kf[sub][0],
                                                (short)0, z, false, false);
    z = __builtin_amdgcn_wmma_f32_16x16x32_bf16(false, qa1, false, kf[sub][1],
                                                (short)0, z, false, false);
    s[sub] = z;
  }

  // scale + causal mask + row-max (reduce across 16-lane half)
  float tmax[8];
#pragma unroll
  for (int r = 0; r < 8; ++r) {
    const int qi = qbase + khalf * 8 + r;
    float v0 = s[0][r] * 0.125f;  // 1/sqrt(64)
    float v1 = s[1][r] * 0.125f;
    if (keybase + l16 > qi)      v0 = -1e30f;
    if (keybase + 16 + l16 > qi) v1 = -1e30f;
    s[0][r] = v0; s[1][r] = v1;
    float t = fmaxf(v0, v1);
    t = fmaxf(t, __shfl_xor(t, 1, 32));
    t = fmaxf(t, __shfl_xor(t, 2, 32));
    t = fmaxf(t, __shfl_xor(t, 4, 32));
    t = fmaxf(t, __shfl_xor(t, 8, 32));
    tmax[r] = t;
  }

  // online softmax update; stage P (bf16) into LDS row-major 16x32
#pragma unroll
  for (int r = 0; r < 8; ++r) {
    const float mnew  = fmaxf(mrow[r], tmax[r]);
    const float alpha = __expf(mrow[r] - mnew);
    const float p0 = __expf(s[0][r] - mnew);
    const float p1 = __expf(s[1][r] - mnew);
    float rs = p0 + p1;
    rs += __shfl_xor(rs, 1, 32);
    rs += __shfl_xor(rs, 2, 32);
    rs += __shfl_xor(rs, 4, 32);
    rs += __shfl_xor(rs, 8, 32);
    lrow[r] = lrow[r] * alpha + rs;
    mrow[r] = mnew;
#pragma unroll
    for (int dt = 0; dt < 4; ++dt) o[dt][r] *= alpha;
    const int row = khalf * 8 + r;
    pw[row * 32 + l16]      = f2bf(p0);
    pw[row * 32 + 16 + l16] = f2bf(p1);
  }

  // LDS RAW: wait on DScnt, then reload P in A-operand layout
  asm volatile("s_wait_dscnt 0" ::: "memory");
  v16bf pa = load_a_frag(pw + l16 * 32, khalf);

  // O += P @ V : 4 WMMAs (one per 16-wide d subtile)
#pragma unroll
  for (int dt = 0; dt < 4; ++dt)
    o[dt] = __builtin_amdgcn_wmma_f32_16x16x32_bf16(
        false, pa, false, vf[dt], (short)0, o[dt], false, false);

  asm volatile("" ::: "memory"); // keep next step's LDS stores after pa load
}

// ---------------- flash attention (causal, online softmax) ----------------
// grid: (T/16/8, B*H), block 256 (8 waves); wave owns one 16-query tile.
// Key-tile loop unrolled by 2 with ping-pong K-fragment buffers (no copies).
__global__ __launch_bounds__(256) void flash_attn_kernel(
    const bf16_t* __restrict__ q, const bf16_t* __restrict__ k,
    const bf16_t* __restrict__ vT, bf16_t* __restrict__ y) {
  __shared__ alignas(16) bf16_t psh[8][16 * 32];  // per-wave P staging (1KB)

  const int lane  = threadIdx.x & 31;
  const int wave  = threadIdx.x >> 5;
  const int khalf = lane >> 4;
  const int l16   = lane & 15;
  const int bh    = blockIdx.y;                 // 0..B*H-1
  const int qtile = blockIdx.x * 8 + wave;      // 0..T/16-1
  const int qbase = qtile * 16;

  const bf16_t* qh = q  + (size_t)bh * Tc * Dc;
  const bf16_t* kh = k  + (size_t)bh * Tc * Dc;
  const bf16_t* vh = vT + (size_t)bh * Dc * Tc;

  // Q tile as two A fragments (K = 0:32 and 32:64)
  const bf16_t* qrow = qh + (size_t)(qbase + l16) * Dc;
  v16bf qa0 = load_a_frag(qrow, khalf);
  v16bf qa1 = load_a_frag(qrow + 32, khalf);

  v8f o[4] = {{}, {}, {}, {}};
  float mrow[8], lrow[8];
#pragma unroll
  for (int r = 0; r < 8; ++r) { mrow[r] = -1e30f; lrow[r] = 0.0f; }

  const int ntiles = (qbase + 15) / 32 + 1;  // causal: only key tiles <= q
  bf16_t* pw = &psh[wave][0];

  v16bf kfA[2][2], kfB[2][2];
  load_k_frags(kh, 0, l16, khalf, kfA);

  int kt = 0;
  for (; kt + 1 < ntiles; kt += 2) {
    // phase 0: consume kfA (tile kt), load kfB (tile kt+1)
    flash_step(kt * 32, (kt + 1) * 32, kh, vh, pw, qbase, l16, khalf,
               qa0, qa1, kfA, kfB, o, mrow, lrow);
    // phase 1: consume kfB (tile kt+1), load kfA (tile kt+2, clamped)
    const int nb = (kt + 2 < ntiles) ? (kt + 2) * 32 : (kt + 1) * 32;
    flash_step((kt + 1) * 32, nb, kh, vh, pw, qbase, l16, khalf,
               qa0, qa1, kfB, kfA, o, mrow, lrow);
  }
  if (kt < ntiles) {  // odd tail: kfA holds tile kt
    flash_step(kt * 32, kt * 32, kh, vh, pw, qbase, l16, khalf,
               qa0, qa1, kfA, kfB, o, mrow, lrow);
  }

  // finalize: O /= l, write y as [B*T, C] bf16 (A operand for out-proj)
  const int bb = bh >> 4;  // / H
  const int h  = bh & (Hc - 1);
  bf16_t* dst = y + (size_t)(bb * Tc + qbase + khalf * 8) * Cc + h * Dc + l16;
#pragma unroll
  for (int r = 0; r < 8; ++r) {
    const float inv = 1.0f / lrow[r];
#pragma unroll
    for (int dt = 0; dt < 4; ++dt)
      dst[r * Cc + dt * 16] = f2bf(o[dt][r] * inv);
  }
}

// ---------------- output projection + bias (fp32 out) ----------------
__global__ __launch_bounds__(256) void out_proj_kernel(
    const bf16_t* __restrict__ yb, const bf16_t* __restrict__ woT,
    const float* __restrict__ bo, float* __restrict__ out) {
  const int lane  = threadIdx.x & 31;
  const int wave  = threadIdx.x >> 5;
  const int khalf = lane >> 4;
  const int l16   = lane & 15;
  const int mbase = (blockIdx.y * 8 + wave) * 32;
  const int nbase = blockIdx.x * 64;

  const bf16_t* arow0 = yb + (size_t)(mbase + l16) * Cc;
  const bf16_t* arow1 = yb + (size_t)(mbase + 16 + l16) * Cc;
  const bf16_t* wrow  = woT + (size_t)(nbase + l16) * Cc;

  v8f acc[2][4] = {{{}, {}, {}, {}}, {{}, {}, {}, {}}};
  gemm_mainloop(arow0, arow1, wrow, khalf, acc);

#pragma unroll
  for (int sub = 0; sub < 2; ++sub) {
#pragma unroll
    for (int nt = 0; nt < 4; ++nt) {
      const int n = nbase + nt * 16 + l16;
      const float bias = bo[n];
      float* dst = out + (size_t)(mbase + sub * 16 + khalf * 8) * Cc + n;
#pragma unroll
      for (int r = 0; r < 8; ++r) dst[r * Cc] = acc[sub][nt][r] + bias;
    }
  }
}

// ---------------- host launcher ----------------
extern "C" void kernel_launch(void* const* d_in, const int* in_sizes, int n_in,
                              void* d_out, int out_size, void* d_ws, size_t ws_size,
                              hipStream_t stream) {
  const float* x  = (const float*)d_in[0];
  const float* Wq = (const float*)d_in[1];
  const float* Wk = (const float*)d_in[2];
  const float* Wv = (const float*)d_in[3];
  const float* Wo = (const float*)d_in[4];
  const float* bo = (const float*)d_in[5];

  char* ws = (char*)d_ws;
  size_t off = 0;
  auto alloc = [&](size_t bytes) -> char* {
    char* p = ws + off;
    off += (bytes + 255) & ~(size_t)255;
    return p;
  };

  bf16_t* xb  = (bf16_t*)alloc((size_t)Mc * Cc * 2);   // x  bf16 [M,C]
  bf16_t* wqT = (bf16_t*)alloc((size_t)Cc * Cc * 2);   // Wq^T bf16
  bf16_t* wkT = (bf16_t*)alloc((size_t)Cc * Cc * 2);
  bf16_t* wvT = (bf16_t*)alloc((size_t)Cc * Cc * 2);
  bf16_t* woT = (bf16_t*)alloc((size_t)Cc * Cc * 2);
  bf16_t* qb  = (bf16_t*)alloc((size_t)Bc * Hc * Tc * Dc * 2); // [B,H,T,D]
  bf16_t* kb  = (bf16_t*)alloc((size_t)Bc * Hc * Tc * Dc * 2); // [B,H,T,D]
  bf16_t* vb  = (bf16_t*)alloc((size_t)Bc * Hc * Dc * Tc * 2); // [B,H,D,T]
  bf16_t* yb  = (bf16_t*)alloc((size_t)Mc * Cc * 2);           // attn out [M,C]

  // 1) conversions
  const int nx = Mc * Cc;
  cvt_f32_to_bf16<<<nx / 4 / 256, 256, 0, stream>>>(x, xb, nx);
  const int nw = Cc * Cc;
  cvt_transpose_bf16<<<nw / 256, 256, 0, stream>>>(Wq, wqT);
  cvt_transpose_bf16<<<nw / 256, 256, 0, stream>>>(Wk, wkT);
  cvt_transpose_bf16<<<nw / 256, 256, 0, stream>>>(Wv, wvT);
  cvt_transpose_bf16<<<nw / 256, 256, 0, stream>>>(Wo, woT);

  // 2) fused QKV projections (WMMA, 32x64 tiles, ping-pong pipelined)
  qkv_gemm_kernel<<<dim3(Cc / 64, Mc / 32 / 8, 3), 256, 0, stream>>>(
      xb, wqT, wkT, wvT, qb, kb, vb);

  // 3) causal flash attention (WMMA, ping-pong K frags, pipelined V frags)
  flash_attn_kernel<<<dim3(Tc / 16 / 8, Bc * Hc), 256, 0, stream>>>(
      qb, kb, vb, yb);

  // 4) output projection + bias (WMMA), fp32 result
  out_proj_kernel<<<dim3(Cc / 64, Mc / 32 / 8), 256, 0, stream>>>(
      yb, woT, bo, (float*)d_out);
}
